// PatchAttention_78675210928472
// MI455X (gfx1250) — matrix operand order repaired and checked by reading, not compiled, not run on hardware
//
#include <hip/hip_runtime.h>

// ---- CDNA5 WMMA / TDM types ----
typedef __bf16 bf16_t;
typedef bf16_t v16bf __attribute__((ext_vector_type(16)));
typedef float  v8f   __attribute__((ext_vector_type(8)));
typedef unsigned int u32x4 __attribute__((ext_vector_type(4)));
typedef int          i32x4 __attribute__((ext_vector_type(4)));
typedef int          i32x8 __attribute__((ext_vector_type(8)));

// fp32 -> bf16 raw bits via native convert (lets backend use v_cvt_*_bf16_f32)
static __device__ __forceinline__ unsigned short f2bf(float f) {
    bf16_t h = (bf16_t)f;
    return __builtin_bit_cast(unsigned short, h);
}

// Problem constants
#define BATCH 8
#define CLOW 128
#define CHIGH 512
#define EMB 256
#define HW 160
#define NH 8
#define CH 16            // channels per head
#define PG 10            // patch grid per dim
#define NP 100           // patches
#define HWSQ 25600       // 160*160
#define WROW 132         // LDS row stride (halfwords) for W, avoids bank conflicts

// ------------------------------------------------------------------
// Kernel 1: 16x16 average pool: x[b,c,160,160] -> xp[b, p, c]
// ------------------------------------------------------------------
__global__ void pa_pool_kernel(const float* __restrict__ x, float* __restrict__ xp) {
    int bp = blockIdx.x;                 // b*100 + p
    int b = bp / NP, p = bp - b * NP;
    int c = threadIdx.x;
    int py = p / PG, px = p - py * PG;
    const float* src = x + ((size_t)(b * CLOW + c) * HW + py * 16) * HW + px * 16;
    float s = 0.f;
#pragma unroll
    for (int i = 0; i < 16; ++i) {
        const float4* r = (const float4*)(src + i * HW);
#pragma unroll
        for (int j = 0; j < 4; ++j) { float4 f = r[j]; s += f.x + f.y + f.z + f.w; }
    }
    xp[(size_t)bp * CLOW + c] = s * (1.f / 256.f);
}

// ------------------------------------------------------------------
// Kernel 2: key projection: kk[bh,p,d] = sum_c xp[b,p,c]*key_w[c, head*32+d]
// ------------------------------------------------------------------
__global__ void pa_key_kernel(const float* __restrict__ xp, const float* __restrict__ key_w,
                              float* __restrict__ kk) {
    int bp = blockIdx.x;
    int b = bp / NP, p = bp - b * NP;
    int e = threadIdx.x;
    __shared__ float row[CLOW];
    if (threadIdx.x < CLOW) row[threadIdx.x] = xp[(size_t)bp * CLOW + threadIdx.x];
    __syncthreads();
    float acc = 0.f;
#pragma unroll 8
    for (int c = 0; c < CLOW; ++c) acc = fmaf(row[c], key_w[c * EMB + e], acc);
    int head = e >> 5, d = e & 31;
    kk[((size_t)(b * NH + head) * NP + p) * 32 + d] = acc;
}

// ------------------------------------------------------------------
// Kernel 3: 3x3 stride-2 pad-1 conv + BN(eval) + SiLU -> qq[bh,p,d]
// ------------------------------------------------------------------
__global__ void pa_conv_kernel(const float* __restrict__ xh, const float* __restrict__ cw,
                               const float* __restrict__ gamma, const float* __restrict__ beta,
                               const float* __restrict__ mean, const float* __restrict__ var,
                               float* __restrict__ qq) {
    const int b = blockIdx.x / PG;
    const int oy = blockIdx.x - b * PG;
    const int e = threadIdx.x;
    __shared__ float win[128 * 3 * 21];   // [ci][ky][21], col 0 == ix=-1 zero pad
    float acc[PG];
#pragma unroll
    for (int i = 0; i < PG; ++i) acc[i] = 0.f;

    for (int cb = 0; cb < 4; ++cb) {
        __syncthreads();
        for (int idx = threadIdx.x; idx < 128 * 63; idx += 256) {
            int ci = idx / 63; int rem = idx - ci * 63;
            int ky = rem / 21; int col = rem - ky * 21;
            int iy = 2 * oy - 1 + ky;
            float v = 0.f;
            if (col > 0 && iy >= 0 && iy < 20)
                v = xh[(((size_t)b * CHIGH + cb * 128 + ci) * 20 + iy) * 20 + (col - 1)];
            win[idx] = v;
        }
        __syncthreads();
        for (int ci = 0; ci < 128; ++ci) {
            const float* wp = cw + ((size_t)e * CHIGH + cb * 128 + ci) * 9;
            float w[9];
#pragma unroll
            for (int j = 0; j < 9; ++j) w[j] = wp[j];
            const float* ls = &win[ci * 63];
#pragma unroll
            for (int ox = 0; ox < PG; ++ox) {
                int ix = 2 * ox;   // padded-row index for kx=0
                float s = acc[ox];
#pragma unroll
                for (int ky = 0; ky < 3; ++ky)
#pragma unroll
                    for (int kx = 0; kx < 3; ++kx)
                        s = fmaf(w[ky * 3 + kx], ls[ky * 21 + ix + kx], s);
                acc[ox] = s;
            }
        }
    }
    float sc = gamma[e] * rsqrtf(var[e] + 1e-5f);
    float sh = beta[e] - mean[e] * sc;
    int head = e >> 5, d = e & 31;
    int bh = b * NH + head;
#pragma unroll
    for (int ox = 0; ox < PG; ++ox) {
        float v = acc[ox] * sc + sh;
        v = v / (1.f + __expf(-v));                 // SiLU
        qq[((size_t)bh * NP + oy * PG + ox) * 32 + d] = v;
    }
}

// ------------------------------------------------------------------
// Kernel 4: scores + softmax, packed to zero-padded bf16 W[64][112][128]
// ------------------------------------------------------------------
__global__ void pa_score_kernel(const float* __restrict__ qq, const float* __restrict__ kk,
                                unsigned short* __restrict__ wtsb) {
    int bh = blockIdx.x;
    __shared__ float qs[NP * 32];
    __shared__ float ks[NP * 32];
    for (int i = threadIdx.x; i < NP * 32; i += 256) {
        qs[i] = qq[(size_t)bh * NP * 32 + i];
        ks[i] = kk[(size_t)bh * NP * 32 + i];
    }
    __syncthreads();
    unsigned short* wout = wtsb + (size_t)bh * 112 * 128;
    int t = threadIdx.x;
    if (t < NP) {
        const float scl = 0.17677669529663687f;     // 1/sqrt(32)
        float qr[32];
#pragma unroll
        for (int d = 0; d < 32; ++d) qr[d] = qs[t * 32 + d];
        float mx = -1e30f;
        for (int j = 0; j < NP; ++j) {
            float s = 0.f;
#pragma unroll
            for (int d = 0; d < 32; ++d) s = fmaf(qr[d], ks[j * 32 + d], s);
            mx = fmaxf(mx, s * scl);
        }
        float sum = 0.f;
        for (int j = 0; j < NP; ++j) {
            float s = 0.f;
#pragma unroll
            for (int d = 0; d < 32; ++d) s = fmaf(qr[d], ks[j * 32 + d], s);
            sum += __expf(s * scl - mx);
        }
        float inv = 1.f / sum;
        for (int j = 0; j < NP; ++j) {
            float s = 0.f;
#pragma unroll
            for (int d = 0; d < 32; ++d) s = fmaf(qr[d], ks[j * 32 + d], s);
            wout[t * 128 + j] = f2bf(__expf(s * scl - mx) * inv);
        }
        for (int j = NP; j < 128; ++j) wout[t * 128 + j] = 0;   // K pad
    } else if (t < 112) {
        for (int j = 0; j < 128; ++j) wout[t * 128 + j] = 0;    // M pad rows
    }
}

// ------------------------------------------------------------------
// Kernel 5: Y[100,4096] = W[100,100] @ V[100,4096] per bh, via bf16 WMMA.
// W (112x128 bf16) is staged into LDS by the Tensor Data Mover with HW
// row padding (2 DWORDs every 64 DWORDs -> 132-halfword LDS rows).
// ------------------------------------------------------------------
__global__ void pa_attn_gemm_kernel(const float* __restrict__ x,
                                    const unsigned short* __restrict__ wtsb,
                                    float* __restrict__ out) {
    const int bh = blockIdx.y;
    const int b = bh >> 3, head = bh & 7;
    const int wave = threadIdx.x >> 5;
    const int lane = threadIdx.x & 31;
    const int ntile = blockIdx.x * 8 + wave;     // 0..255
    const int cih = ntile >> 4;                  // channel within head
    const int ph  = ntile & 15;                  // row within patch

    __shared__ __align__(16) unsigned short Wlds[112 * WROW];

    // ---- TDM: stage W tile (128 x 112 bf16, row-padded) global -> LDS ----
    if (wave == 0) {
        unsigned int ldsa = (unsigned int)(unsigned long long)&Wlds[0];   // LDS byte offset
        unsigned long long ga = (unsigned long long)(const void*)(wtsb + (size_t)bh * 112 * 128);
        // D# group 0: count=1 | lds_addr | global_addr | type=2
        u32x4 g0 = { 1u, ldsa, (unsigned int)ga,
                     ((unsigned int)((ga >> 32) & 0x1FFFFFFull)) | 0x80000000u };
        // D# group 1: data_size=2B, pad_enable, pad_interval=64 DW, pad_amount=2 DW,
        //             tensor_dim0=128, tensor_dim1=112, tile_dim0=128, tile_dim1=112,
        //             tensor_dim0_stride=128
        i32x8 g1 = { 0x03510000, 0x00800000, 0x00700000, 0x00800000,
                     112, 128, 0, 0 };
        i32x4 z4 = { 0, 0, 0, 0 };
#if defined(__clang_major__) && (__clang_major__ >= 23)
        i32x8 z8 = { 0, 0, 0, 0, 0, 0, 0, 0 };
        __builtin_amdgcn_tensor_load_to_lds(g0, g1, z4, z4, z8, 0);
#else
        __builtin_amdgcn_tensor_load_to_lds(g0, g1, z4, z4, 0);
#endif
    }

    // B fragments: lane%16 = N (pw), slot s = K in table order (8*half + s, 16 + 8*half + s-8).
    const float* xb = x + ((size_t)b * CLOW + head * CH + cih) * HWSQ;
    v16bf bfrag[4];
#pragma unroll
    for (int kt = 0; kt < 4; ++kt) {
        v16bf bv;
        const int nB = lane & 15, hB = lane >> 4;
#pragma unroll
        for (int s = 0; s < 16; ++s) {
            int k = kt * 32 + (s < 8 ? hB * 8 + s : 16 + hB * 8 + (s - 8));
            float v = 0.f;
            if (k < NP) {
                int pyk = k / PG, pxk = k - pyk * PG;
                v = xb[(pyk * 16 + ph) * HW + pxk * 16 + nB];
            }
            bv[s] = (bf16_t)v;
        }
        bfrag[kt] = bv;
    }

    // Wait for TDM completion (TENSORcnt is per-wave: only wave 0 issued it),
    // then block-wide barrier so every wave sees the staged W.
    if (wave == 0) __builtin_amdgcn_s_wait_tensorcnt(0);
    __syncthreads();

    const int m = lane & 15, half = lane >> 4;
    float* ob = out + ((size_t)b * CLOW + head * CH + cih) * HWSQ;
    const int n = lane & 15;

    for (int mt = 0; mt < 7; ++mt) {
        v8f acc = {0.f, 0.f, 0.f, 0.f, 0.f, 0.f, 0.f, 0.f};
#pragma unroll
        for (int kt = 0; kt < 4; ++kt) {
            // A fragment (16-bit A 16x32 layout): slots 2v,2v+1 pack K = (v/4)*16 + half*8 + (v%4)*2 (+kt*32)
            union AU { unsigned int u[8]; v16bf v; } af;
            const unsigned short* wr = &Wlds[(mt * 16 + m) * WROW + kt * 32 + half * 8];
            af.u[0] = *(const unsigned int*)(wr + 0);
            af.u[1] = *(const unsigned int*)(wr + 2);
            af.u[2] = *(const unsigned int*)(wr + 4);
            af.u[3] = *(const unsigned int*)(wr + 6);
            af.u[4] = *(const unsigned int*)(wr + 16);
            af.u[5] = *(const unsigned int*)(wr + 18);
            af.u[6] = *(const unsigned int*)(wr + 20);
            af.u[7] = *(const unsigned int*)(wr + 22);
            acc = __builtin_amdgcn_wmma_f32_16x16x32_bf16(
                false, af.v, false, bfrag[kt], (short)0, acc, false, false);
        }
        // Store: lane holds rows M = r + 8*half at column N = lane%16 (pw)
#pragma unroll
        for (int r = 0; r < 8; ++r) {
            int q = mt * 16 + r + half * 8;
            if (q < NP) {
                int pyq = q / PG, pxq = q - pyq * PG;
                ob[(pyq * 16 + ph) * HW + pxq * 16 + n] = acc[r];
            }
        }
    }
}

// ------------------------------------------------------------------
extern "C" void kernel_launch(void* const* d_in, const int* in_sizes, int n_in,
                              void* d_out, int out_size, void* d_ws, size_t ws_size,
                              hipStream_t stream) {
    const float* x      = (const float*)d_in[0];
    const float* x_high = (const float*)d_in[1];
    const float* key_w  = (const float*)d_in[2];
    const float* conv_w = (const float*)d_in[3];
    const float* gamma  = (const float*)d_in[4];
    const float* beta   = (const float*)d_in[5];
    const float* mean   = (const float*)d_in[6];
    const float* var    = (const float*)d_in[7];
    float* out = (float*)d_out;

    char* ws = (char*)d_ws;
    float* xp            = (float*)(ws);                    //  8*100*128 f32 = 409600 B
    float* kk            = (float*)(ws + 409600);           // 64*100*32  f32 = 819200 B
    float* qq            = (float*)(ws + 1228800);          // 64*100*32  f32 = 819200 B
    unsigned short* wtsb = (unsigned short*)(ws + 2048000); // 64*112*128 bf16 = 1835008 B

    pa_pool_kernel <<<BATCH * NP, 128, 0, stream>>>(x, xp);
    pa_key_kernel  <<<BATCH * NP, 256, 0, stream>>>(xp, key_w, kk);
    pa_conv_kernel <<<BATCH * PG, 256, 0, stream>>>(x_high, conv_w, gamma, beta, mean, var, qq);
    pa_score_kernel<<<BATCH * NH, 256, 0, stream>>>(qq, kk, wtsb);
    pa_attn_gemm_kernel<<<dim3(32, BATCH * NH), 256, 0, stream>>>(x, wtsb, out);
}
